// GroupLowRankAttention_2018634629351
// MI455X (gfx1250) — matrix operand order repaired and checked by reading, not compiled, not run on hardware
//
#include <hip/hip_runtime.h>
#include <hip/hip_bf16.h>
#include <math.h>

// ---------------------------------------------------------------------------
// CDNA5 (gfx1250) wave32 BF16 WMMA implementation of GroupLowRankAttention.
// D = A*B+C via v_wmma_f32_16x16x32_bf16; f32 accumulation everywhere.
// LDS GEMM is double-buffered; A-tile staging uses the gfx1250 async
// global->LDS path (ASYNCcnt) when the toolchain exposes the builtin.
// ---------------------------------------------------------------------------

typedef __bf16 bf16_t;
typedef __attribute__((ext_vector_type(16))) __bf16 v16bf;
typedef __attribute__((ext_vector_type(8)))  float  v8f;
typedef int v4i __attribute__((vector_size(16)));   // native vector, as the builtin expects

union Frag16 { v16bf v; uint4 q[2]; };

#ifndef __has_builtin
#define __has_builtin(x) 0
#endif

#if __has_builtin(__builtin_amdgcn_global_load_async_to_lds_b128) && \
    __has_builtin(__builtin_amdgcn_s_wait_asynccnt)
#define USE_ASYNC_LDS 1
#else
#define USE_ASYNC_LDS 0
#endif

#if USE_ASYNC_LDS
#define AS1 __attribute__((address_space(1)))
#define AS3 __attribute__((address_space(3)))
// 16-byte async copy global -> LDS (per-lane addresses), tracked by ASYNCcnt.
__device__ __forceinline__ void async_cp16(void* lds, const void* g)
{
    __builtin_amdgcn_global_load_async_to_lds_b128((AS1 v4i*)g, (AS3 v4i*)lds, 0, 0);
}
__device__ __forceinline__ void async_wait0() { __builtin_amdgcn_s_wait_asynccnt(0); }
#else
__device__ __forceinline__ void async_wait0() {}
#endif

// ---------------------------------------------------------------------------
// Generic 128x128-tile GEMM: C(MxN) = A(MxK,row-major,bf16) * B(KxN,row-major)
//   B_F32:   B source is f32 (converted to bf16 while staging to LDS)
//   EPI = 0: store bf16 C
//   EPI = 1: store bf16 C + accumulate per-row sum-of-squares (L2 norm accum)
//   EPI = 2: store f32 C (non-temporal; write-once final output)
// Block = 256 threads = 8 waves; wave w computes a 16x128 strip.
// Double-buffered LDS: stage tile i+1 (async where possible) while WMMAs run
// on tile i; one barrier per k-step.
// ---------------------------------------------------------------------------
template<int EPI, bool B_F32>
__global__ __launch_bounds__(256)
void gemm128_bf16(const bf16_t* __restrict__ A, long long strideA, int lda,
                  const void* __restrict__ Bsrc, long long strideB, int ldb,
                  void* __restrict__ Cdst, long long strideC, int ldc,
                  int K,
                  float* __restrict__ normOut, int normStride)
{
    __shared__ bf16_t As[2][128][40];   // [m][k], +8 pad keeps 16B-aligned rows
    __shared__ bf16_t Bs[2][128][40];   // [n][k] (transposed) -> contiguous B frags

    const int t  = threadIdx.x;
    const int m0 = blockIdx.y * 128;
    const int n0 = blockIdx.x * 128;
    const size_t bz = blockIdx.z;

    A += bz * (size_t)strideA;
    const char* Bbase = (const char*)Bsrc + bz * (size_t)strideB * (B_F32 ? 4 : 2);

    // staging maps
    const int am  = t >> 1;          // A: 2 threads/row, 16 bf16 each
    const int ak  = (t & 1) * 16;
    const int bn  = t & 127;         // B: 2 threads/column, 16 k-values each
    const int bkh = (t >> 7) * 16;

    // compute maps (wave32)
    const int lane = t & 31;
    const int w    = t >> 5;
    const int lrow = lane & 15;
    const int lhi  = lane >> 4;
    const int mw   = w * 16;

    auto stageA = [&](int k0, int buf) {
        const bf16_t* ap = A + (size_t)(m0 + am) * lda + (k0 + ak);
#if USE_ASYNC_LDS
        async_cp16(&As[buf][am][ak],     ap);
        async_cp16(&As[buf][am][ak + 8], ap + 8);
#else
        uint4 r0 = *(const uint4*)(ap);
        uint4 r1 = *(const uint4*)(ap + 8);
        *(uint4*)&As[buf][am][ak]     = r0;
        *(uint4*)&As[buf][am][ak + 8] = r1;
#endif
    };

    auto stageB = [&](int k0, int buf) {
        alignas(16) bf16_t tmp[16];
        if (B_F32) {
            const float* bp = (const float*)Bbase + (size_t)(k0 + bkh) * ldb + (n0 + bn);
            #pragma unroll
            for (int i = 0; i < 16; ++i) tmp[i] = (bf16_t)bp[(size_t)i * ldb];
            // prefetch the next k-step's rows of the big f32 operand
            __builtin_prefetch(bp + (size_t)32 * ldb, 0, 0);
        } else {
            const bf16_t* bp = (const bf16_t*)Bbase + (size_t)(k0 + bkh) * ldb + (n0 + bn);
            #pragma unroll
            for (int i = 0; i < 16; ++i) tmp[i] = bp[(size_t)i * ldb];
        }
        *(uint4*)&Bs[buf][bn][bkh]     = *(const uint4*)&tmp[0];
        *(uint4*)&Bs[buf][bn][bkh + 8] = *(const uint4*)&tmp[8];
    };

    v8f acc[8];
    #pragma unroll
    for (int i = 0; i < 8; ++i)
        #pragma unroll
        for (int e = 0; e < 8; ++e) acc[i][e] = 0.0f;

    stageA(0, 0);
    stageB(0, 0);

    int cur = 0;
    for (int k0 = 0; k0 < K; k0 += 32, cur ^= 1) {
        async_wait0();        // our async LDS writes (if any) have landed
        __syncthreads();      // everyone's staging for `cur` is visible

        if (k0 + 32 < K) {    // overlap next-tile staging with WMMA on `cur`
            stageA(k0 + 32, cur ^ 1);
            stageB(k0 + 32, cur ^ 1);
        }

        // A fragment: lane = row M (lrow); k = kb..kb+7, kb+16..kb+23 (kb = lhi*8)
        Frag16 a;
        a.q[0] = *(const uint4*)&As[cur][mw + lrow][lhi * 8];
        a.q[1] = *(const uint4*)&As[cur][mw + lrow][lhi * 8 + 16];

        #pragma unroll
        for (int tn = 0; tn < 8; ++tn) {
            // B fragment: lane = col N (lrow); k = kb..kb+7, kb+16..kb+23 (kb = lhi*8)
            Frag16 b;
            b.q[0] = *(const uint4*)&Bs[cur][tn * 16 + lrow][lhi * 8];
            b.q[1] = *(const uint4*)&Bs[cur][tn * 16 + lrow][lhi * 8 + 16];
            acc[tn] = __builtin_amdgcn_wmma_f32_16x16x32_bf16(
                false, a.v, false, b.v, (short)0, acc[tn], false, false);
        }
    }

    // ---- epilogue: C/D layout M = v + 8*lhi, N = lrow ----
    if (EPI == 2) {
        float* C = (float*)((char*)Cdst + bz * (size_t)strideC * 4);
        #pragma unroll
        for (int v = 0; v < 8; ++v) {
            const int grow = m0 + mw + v + lhi * 8;
            #pragma unroll
            for (int tn = 0; tn < 8; ++tn)
                __builtin_nontemporal_store(
                    acc[tn][v], &C[(size_t)grow * ldc + (n0 + tn * 16 + lrow)]);
        }
    } else {
        bf16_t* C = (bf16_t*)((char*)Cdst + bz * (size_t)strideC * 2);
        #pragma unroll
        for (int v = 0; v < 8; ++v) {
            const int grow = m0 + mw + v + lhi * 8;
            #pragma unroll
            for (int tn = 0; tn < 8; ++tn)
                C[(size_t)grow * ldc + (n0 + tn * 16 + lrow)] = (bf16_t)acc[tn][v];
        }
        if (EPI == 1) {
            float* nb = normOut + bz * (size_t)normStride;
            #pragma unroll
            for (int v = 0; v < 8; ++v) {
                float s = 0.0f;
                #pragma unroll
                for (int tn = 0; tn < 8; ++tn) { float x = acc[tn][v]; s += x * x; }
                // reduce across the 16 lanes sharing this row (xor<16 stays in half)
                s += __shfl_xor(s, 1);
                s += __shfl_xor(s, 2);
                s += __shfl_xor(s, 4);
                s += __shfl_xor(s, 8);
                if (lrow == 0) atomicAdd(&nb[m0 + mw + v + lhi * 8], s);
            }
        }
    }
}

// ---------------------------------------------------------------------------
// att[b] = normalize(Qr) * normalize(Kr)^T * temp   (M=N=r=256, K=Nspatial)
// Both operands are k-contiguous per lane -> fragments loaded straight from
// global as b128s (no LDS staging needed). Scores stored f32.
// Grid: (r/64, 1, B); wave w covers rows [w*32, w*32+32), block covers 64 cols.
// ---------------------------------------------------------------------------
__global__ __launch_bounds__(256)
void att_scores(const bf16_t* __restrict__ Qr, const bf16_t* __restrict__ Kr,
                const float* __restrict__ normQ, const float* __restrict__ normK,
                const float* __restrict__ temp,
                float* __restrict__ attOut, int N, int r)
{
    const int t    = threadIdx.x;
    const int lane = t & 31, w = t >> 5;
    const int lrow = lane & 15, lhi = lane >> 4;
    const size_t b = blockIdx.z;
    const int n0   = blockIdx.x * 64;
    const int mw   = w * 32;

    const bf16_t* Qb = Qr + b * (size_t)r * N;
    const bf16_t* Kb = Kr + b * (size_t)r * N;

    v8f acc[2][4];
    #pragma unroll
    for (int i = 0; i < 2; ++i)
        #pragma unroll
        for (int j = 0; j < 4; ++j)
            #pragma unroll
            for (int e = 0; e < 8; ++e) acc[i][j][e] = 0.0f;

    for (int k0 = 0; k0 < N; k0 += 32) {
        Frag16 aq[2];
        #pragma unroll
        for (int tm = 0; tm < 2; ++tm) {
            const bf16_t* p = Qb + (size_t)(mw + tm * 16 + lrow) * N + k0 + lhi * 8;
            aq[tm].q[0] = *(const uint4*)p;
            aq[tm].q[1] = *(const uint4*)(p + 16);
        }
        #pragma unroll
        for (int tn = 0; tn < 4; ++tn) {
            Frag16 bk;  // B[k][j] = Kr[j][k]: contiguous k per lane
            const bf16_t* p = Kb + (size_t)(n0 + tn * 16 + lrow) * N + k0 + lhi * 8;
            bk.q[0] = *(const uint4*)p;
            bk.q[1] = *(const uint4*)(p + 16);
            #pragma unroll
            for (int tm = 0; tm < 2; ++tm)
                acc[tm][tn] = __builtin_amdgcn_wmma_f32_16x16x32_bf16(
                    false, aq[tm].v, false, bk.v, (short)0, acc[tm][tn], false, false);
        }
    }

    const float tv = temp[0];
    #pragma unroll
    for (int tm = 0; tm < 2; ++tm) {
        #pragma unroll
        for (int v = 0; v < 8; ++v) {
            const int row = mw + tm * 16 + v + lhi * 8;
            const float sq = 1.0f / fmaxf(sqrtf(normQ[b * r + row]), 1e-12f);
            #pragma unroll
            for (int tn = 0; tn < 4; ++tn) {
                const int col = n0 + tn * 16 + lrow;
                const float sk = 1.0f / fmaxf(sqrtf(normK[b * r + col]), 1e-12f);
                attOut[(b * r + row) * (size_t)r + col] = acc[tm][tn][v] * sq * sk * tv;
            }
        }
    }
}

// ---------------------------------------------------------------------------
// Row softmax over r=256 (f32 in, bf16 probabilities out). One wave per row.
// ---------------------------------------------------------------------------
__global__ __launch_bounds__(256)
void softmax_rows(const float* __restrict__ att, bf16_t* __restrict__ attP, int r)
{
    const int lane  = threadIdx.x & 31;
    const size_t row = (size_t)blockIdx.x * 8 + (threadIdx.x >> 5);
    const float* src = att + row * r;

    float x[8];
    float m = -3.4e38f;
    #pragma unroll
    for (int i = 0; i < 8; ++i) { x[i] = src[lane + i * 32]; m = fmaxf(m, x[i]); }
    #pragma unroll
    for (int d = 16; d >= 1; d >>= 1) m = fmaxf(m, __shfl_xor(m, d));
    float s = 0.0f;
    #pragma unroll
    for (int i = 0; i < 8; ++i) { x[i] = __expf(x[i] - m); s += x[i]; }
    #pragma unroll
    for (int d = 16; d >= 1; d >>= 1) s += __shfl_xor(s, d);
    const float inv = 1.0f / s;
    bf16_t* dst = attP + row * r;
    #pragma unroll
    for (int i = 0; i < 8; ++i) dst[lane + i * 32] = (bf16_t)(x[i] * inv);
}

__global__ void cvt_f32_bf16(const float* __restrict__ src, bf16_t* __restrict__ dst, size_t n)
{
    size_t i = (size_t)blockIdx.x * blockDim.x + threadIdx.x;
    if (i < n) dst[i] = (bf16_t)src[i];
}

// ---------------------------------------------------------------------------
extern "C" void kernel_launch(void* const* d_in, const int* in_sizes, int n_in,
                              void* d_out, int out_size, void* d_ws, size_t ws_size,
                              hipStream_t stream)
{
    const float* qg   = (const float*)d_in[0];
    const float* kg   = (const float*)d_in[1];
    const float* vg   = (const float*)d_in[2];
    const float* temp = (const float*)d_in[3];
    const float* Wq   = (const float*)d_in[4];
    const float* Wk   = (const float*)d_in[5];
    const float* Wv   = (const float*)d_in[6];
    const float* Wb   = (const float*)d_in[7];
    float* out = (float*)d_out;
    (void)in_sizes; (void)n_in; (void)out_size; (void)ws_size;

    const int B = 16, Cg = 1024, N = 4096, r = 256;

    char* p = (char*)d_ws;
    auto carve = [&](size_t bytes) -> void* {
        void* q = (void*)p;
        p += (bytes + 255) & ~(size_t)255;
        return q;
    };
    bf16_t* Wq_b = (bf16_t*)carve((size_t)r * Cg * 2);
    bf16_t* Wk_b = (bf16_t*)carve((size_t)r * Cg * 2);
    bf16_t* Wv_b = (bf16_t*)carve((size_t)r * Cg * 2);
    bf16_t* Wb_b = (bf16_t*)carve((size_t)Cg * r * 2);
    bf16_t* Qr   = (bf16_t*)carve((size_t)B * r * N * 2);
    bf16_t* Kr   = (bf16_t*)carve((size_t)B * r * N * 2);
    bf16_t* Vr   = (bf16_t*)carve((size_t)B * r * N * 2);
    float*  nQ   = (float*)carve((size_t)B * r * 4);
    float*  nK   = (float*)carve((size_t)B * r * 4);
    float*  attS = (float*)carve((size_t)B * r * r * 4);
    bf16_t* attP = (bf16_t*)carve((size_t)B * r * r * 2);
    bf16_t* outR = (bf16_t*)carve((size_t)B * r * N * 2);

    dim3 blk(256);

    // weights -> bf16 once; zero L2-norm accumulators
    cvt_f32_bf16<<<(r * Cg + 255) / 256, blk, 0, stream>>>(Wq, Wq_b, (size_t)r * Cg);
    cvt_f32_bf16<<<(r * Cg + 255) / 256, blk, 0, stream>>>(Wk, Wk_b, (size_t)r * Cg);
    cvt_f32_bf16<<<(r * Cg + 255) / 256, blk, 0, stream>>>(Wv, Wv_b, (size_t)r * Cg);
    cvt_f32_bf16<<<(Cg * r + 255) / 256, blk, 0, stream>>>(Wb, Wb_b, (size_t)Cg * r);
    (void)hipMemsetAsync(nQ, 0, (size_t)B * r * 4, stream);
    (void)hipMemsetAsync(nK, 0, (size_t)B * r * 4, stream);

    // 1) projections: P[b](r x N) = W(r x Cg) * X[b](Cg x N)   (X is f32)
    dim3 gProj(N / 128, r / 128, B);
    gemm128_bf16<1, true><<<gProj, blk, 0, stream>>>(
        Wq_b, 0, Cg, qg, (long long)Cg * N, N, Qr, (long long)r * N, N, Cg, nQ, r);
    gemm128_bf16<1, true><<<gProj, blk, 0, stream>>>(
        Wk_b, 0, Cg, kg, (long long)Cg * N, N, Kr, (long long)r * N, N, Cg, nK, r);
    gemm128_bf16<0, true><<<gProj, blk, 0, stream>>>(
        Wv_b, 0, Cg, vg, (long long)Cg * N, N, Vr, (long long)r * N, N, Cg, nullptr, 0);

    // 2) attention scores (normalization + temp folded into epilogue)
    att_scores<<<dim3(r / 64, 1, B), blk, 0, stream>>>(Qr, Kr, nQ, nK, temp, attS, N, r);

    // 3) softmax rows -> bf16 probabilities
    softmax_rows<<<dim3((B * r) / 8), blk, 0, stream>>>(attS, attP, r);

    // 4) out_r[b](r x N) = att[b](r x r) * Vr[b](r x N)
    gemm128_bf16<0, false><<<dim3(N / 128, r / 128, B), blk, 0, stream>>>(
        attP, (long long)r * r, r, Vr, (long long)r * N, N, outR, (long long)r * N, N, r, nullptr, 0);

    // 5) out[b](Cg x N) = Wb(Cg x r) * out_r[b](r x N)   (f32 output, NT stores)
    gemm128_bf16<2, false><<<dim3(N / 128, Cg / 128, B), blk, 0, stream>>>(
        Wb_b, 0, r, outR, (long long)r * N, N, out, (long long)Cg * N, N, r, nullptr, 0);
}
